// E3nnMACE_33509334843736
// MI455X (gfx1250) — matrix operand order repaired and checked by reading, not compiled, not run on hardware
//
#include <hip/hip_runtime.h>
#include <hip/hip_bf16.h>
#include <math.h>

// ---------------------------------------------------------------------------
// E3nn/MACE interaction on MI455X (gfx1250, wave32, WMMA).
// Channel mixing recast as two edge-shared GEMMs: out = elen*(z^T W) + (z^T b),
// executed with v_wmma_f32_16x16x32_bf16 chains (f32 accumulate).
// ---------------------------------------------------------------------------

typedef __attribute__((ext_vector_type(16))) __bf16 v16bf;
typedef __attribute__((ext_vector_type(8)))  float  v8f;

#define NPATH   23
#define CG_CAP  360           // per-path scratch capacity in prep
#define CG_TOT  1024          // compacted total capacity (true bound: <=878)
#define CH      32            // MUL (channels)
#define NDIM    512           // NODE_DIM
#define TILE_E  8             // edges per block in the interaction kernel

// ---- tables derived from the reference's _build_paths() --------------------
__constant__ int PL1[NPATH]  = {0,0,0,0, 1,1,1,1,1,1, 2,2,2,2,2,2,2, 3,3,3,3,3,3};
__constant__ int PL2[NPATH]  = {0,1,2,3, 0,1,1,2,2,3, 0,1,1,2,2,3,3, 0,1,2,2,3,3};
__constant__ int PL3[NPATH]  = {0,1,2,3, 1,0,2,1,3,2, 2,1,3,0,2,1,3, 3,2,1,3,0,2};
__constant__ int PSLOT[NPATH]= {0,0,0,0, 1,1,1,2,1,2, 3,3,2,2,4,4,3, 4,5,5,5,3,6};
__constant__ int XLOC[4]  = {0, 1, 4, 9};
__constant__ int NOUTC[4] = {4, 6, 7, 6};
__constant__ int KL3c[4]  = {128, 192, 224, 192};
__constant__ int ZOFFc[4] = {0, 1024, 5632, 14592};
__constant__ int PATH_OF_KT[NPATH] = {0,5,13,21, 1,4,7,11,15,19, 2,6,9,10,14,18,22, 3,8,12,16,17,20};
// 20 GEMM jobs: (l3, m-tile, n-half)
__constant__ int JOB_L3[20] = {0,0, 1,1,1,1, 2,2,2,2,2,2, 3,3,3,3,3,3,3,3};
__constant__ int JOB_MT[20] = {0,0, 0,0,1,1, 0,0,1,1,2,2, 0,0,1,1,2,2,3,3};
__constant__ int JOB_NH[20] = {0,1, 0,1,0,1, 0,1,0,1,0,1, 0,1,0,1,0,1,0,1};

// compile-time copies for the templated GEMM job
constexpr int cKT[4]    = {4, 6, 7, 6};
constexpr int cKTOFF[4] = {0, 4, 10, 17};
constexpr int cKL3[4]   = {128, 192, 224, 192};
constexpr int cZOFF[4]  = {0, 1024, 5632, 14592};
constexpr int cNOFF[4]  = {0, 32, 128, 288};

struct CgEnt { unsigned char ia, ja, kk, pad; float v; };   // 8 bytes

__device__ __forceinline__ unsigned short f2bf(float x) {   // RNE f32 -> bf16
    unsigned int u = __float_as_uint(x);
    u += 0x7FFFu + ((u >> 16) & 1u);
    return (unsigned short)(u >> 16);
}
__device__ __forceinline__ int imax2(int a, int b){ return a > b ? a : b; }
__device__ __forceinline__ int imin2(int a, int b){ return a < b ? a : b; }

// ---------------------------------------------------------------------------
// Kernel 0: real-CG nonzero tables on device (exact port of reference math),
// compacted into one contiguous stream for LDS staging.
// hdr layout: hdr[2p]=offset, hdr[2p+1]=count, hdr[46]=total.
// ---------------------------------------------------------------------------
__device__ double dfact(int n) { double r = 1.0; for (int i = 2; i <= n; i++) r *= (double)i; return r; }

__device__ double su2_cg(int j1,int j2,int j3,int m1,int m2,int m3) {
    if (m1 + m2 != m3) return 0.0;
    if (j3 < abs(j1 - j2) || j3 > j1 + j2) return 0.0;
    if (abs(m1) > j1 || abs(m2) > j2 || abs(m3) > j3) return 0.0;
    double pref = sqrt((2.0*j3 + 1.0) * dfact(j3+j1-j2) * dfact(j3-j1+j2) * dfact(j1+j2-j3) / dfact(j1+j2+j3+1));
    pref *= sqrt(dfact(j3+m3)*dfact(j3-m3)*dfact(j1-m1)*dfact(j1+m1)*dfact(j2-m2)*dfact(j2+m2));
    double s = 0.0;
    int kmin = imax2(0, imax2(j2 - j3 - m1, j1 - j3 + m2));
    int kmax = imin2(j1 + j2 - j3, imin2(j1 - m1, j2 + m2));
    for (int k = kmin; k <= kmax; k++) {
        double d = dfact(k)*dfact(j1+j2-j3-k)*dfact(j1-m1-k)*dfact(j2+m2-k)*dfact(j3-j2+m1+k)*dfact(j3-j1-m2+k);
        s += ((k & 1) ? -1.0 : 1.0) / d;
    }
    return pref * s;
}

__device__ void c2r_fill(int l, double Ur[7][7], double Ui[7][7]) {
    for (int i = 0; i < 7; i++) for (int j = 0; j < 7; j++) { Ur[i][j] = 0.0; Ui[i][j] = 0.0; }
    double inv2 = 1.0 / sqrt(2.0);
    for (int m = -l; m <= l; m++) {
        int i = m + l;
        if (m > 0) { Ur[i][l-m] = inv2; Ur[i][l+m] = ((m & 1) ? -1.0 : 1.0) * inv2; }
        else if (m == 0) { Ur[i][l] = 1.0; }
        else { Ui[i][l+m] = inv2; Ui[i][l-m] = -(((-m) & 1) ? -1.0 : 1.0) * inv2; }
    }
}

__global__ void cg_prep_kernel(int* __restrict__ hdr, CgEnt* __restrict__ scratch,
                               CgEnt* __restrict__ compact) {
    __shared__ int cnts[NPATH];
    __shared__ int offs[NPATH + 1];
    int p = threadIdx.x;
    if (p < NPATH) {
        int l1 = PL1[p], l2 = PL2[p], l3 = PL3[p];
        int d1 = 2*l1 + 1, d2 = 2*l2 + 1, d3 = 2*l3 + 1;
        double U1r[7][7], U1i[7][7], U2r[7][7], U2i[7][7], U3r[7][7], U3i[7][7];
        c2r_fill(l1, U1r, U1i); c2r_fill(l2, U2r, U2i); c2r_fill(l3, U3r, U3i);
        double C[7][7][7];
        for (int a = 0; a < d1; a++) for (int b = 0; b < d2; b++) for (int c = 0; c < d3; c++)
            C[a][b][c] = su2_cg(l1, l2, l3, a - l1, b - l2, c - l3);
        double norm = 1.0 / sqrt((double)CH * (double)NOUTC[l3]);
        int cnt = 0;
        CgEnt* dst = scratch + (size_t)p * CG_CAP;
        for (int i = 0; i < d1; i++) for (int j = 0; j < d2; j++) for (int k = 0; k < d3; k++) {
            double sr = 0.0;
            for (int a = 0; a < d1; a++) for (int b = 0; b < d2; b++) {
                double tr = U1r[i][a]*U2r[j][b] - U1i[i][a]*U2i[j][b];
                double ti = U1r[i][a]*U2i[j][b] + U1i[i][a]*U2r[j][b];
                for (int c = 0; c < d3; c++) {
                    double Cv = C[a][b][c];
                    if (Cv == 0.0) continue;
                    sr += (tr * U3r[k][c] + ti * U3i[k][c]) * Cv;  // Re(U1*U2*conj(U3))*C
                }
            }
            float val = (float)(sr * norm);
            if (fabsf(val) > 1e-7f && cnt < CG_CAP) {
                CgEnt e; e.ia = (unsigned char)(XLOC[l1] + i); e.ja = (unsigned char)(XLOC[l2] + j);
                e.kk = (unsigned char)k; e.pad = 0; e.v = val;
                dst[cnt++] = e;
            }
        }
        cnts[p] = cnt;
    }
    __syncthreads();
    if (threadIdx.x == 0) {                 // serial prefix over 23 entries
        int acc = 0;
        for (int q = 0; q < NPATH; q++) { offs[q] = acc; acc += cnts[q]; }
        offs[NPATH] = acc <= CG_TOT ? acc : CG_TOT;
        for (int q = 0; q < NPATH; q++) { hdr[2*q] = offs[q]; hdr[2*q+1] = cnts[q]; }
        hdr[2*NPATH] = offs[NPATH];
    }
    __syncthreads();
    if (p < NPATH) {                        // copy into compact stream
        int base = offs[p], cnt = cnts[p];
        const CgEnt* s = scratch + (size_t)p * CG_CAP;
        for (int t = 0; t < cnt; t++)
            if (base + t < CG_TOT) compact[base + t] = s[t];
    }
}

// ---------------------------------------------------------------------------
// Kernel 1: swizzle weights into WMMA B-fragment layout (bf16)
// wfrag[layer][ktG(23)][set(2:W,b)][nhalf(2)][lane(32)][16 halves]
// B (32x16) 16-bit layout: element s of lane half hi holds
// klocal = 8*hi + s (s<8), 16 + 8*hi + (s-8) (s>=8).
// ---------------------------------------------------------------------------
#define WFRAG_LAYER 47104     // ushorts per layer: 23*2*2*32*16
__global__ void w_prep_kernel(const float* __restrict__ wnet_w, const float* __restrict__ wnet_b,
                              unsigned short* __restrict__ wfrag) {
    int gid = blockIdx.x * blockDim.x + threadIdx.x;
    if (gid >= 2 * NPATH * 2 * 2 * 32) return;      // 5888
    int lane  = gid & 31;
    int nh    = (gid >> 5) & 1;
    int set   = (gid >> 6) & 1;
    int ktG   = (gid >> 7) % NPATH;
    int layer = (gid >> 7) / NPATH;
    int path  = PATH_OF_KT[ktG];
    const float* src = (set == 0 ? wnet_w : wnet_b) + (size_t)layer * 23552 + (size_t)path * 1024;
    int n  = (lane & 15) + 16 * nh;
    int hi = lane >> 4;
    unsigned short* out = wfrag + (size_t)layer * WFRAG_LAYER
                        + ((((size_t)ktG * 2 + set) * 2 + nh) * 32 + lane) * 16;
    #pragma unroll
    for (int h = 0; h < 16; h++) {
        int r = h >> 1, hh = h & 1;
        int kl = (h < 8) ? (8 * hi + 2 * r + hh) : (16 + 8 * hi + 2 * r + hh - 8);   // u (source channel)
        out[h] = f2bf(src[kl * 32 + n]);
    }
}

// ---------------------------------------------------------------------------
// Kernel 2: per-edge geometry + spherical harmonics
// ---------------------------------------------------------------------------
__global__ void geom_kernel(const float* __restrict__ pos, const int* __restrict__ ei, int E,
                            float* __restrict__ elen, float* __restrict__ Yb) {
    int e = blockIdx.x * blockDim.x + threadIdx.x;
    if (e >= E) return;
    int s = ei[e], d = ei[E + e];
    float ex = pos[3*d+0] - pos[3*s+0];
    float ey = pos[3*d+1] - pos[3*s+1];
    float ez = pos[3*d+2] - pos[3*s+2];
    float len = fmaxf(sqrtf(ex*ex + ey*ey + ez*ez), 1e-8f);
    float x = ex / len, y = ey / len, z = ez / len;
    const float s3 = 1.7320508075688772f, s5 = 2.2360679774997896f, s7 = 2.6457513110645907f;
    const float s15 = 3.872983346207417f, s42 = 6.48074069840786f, s70 = 8.366600265340756f;
    const float s105 = 10.246950765959598f;
    float* Y = Yb + (size_t)e * 16;
    Y[0] = 1.0f;
    Y[1] = s3*y;  Y[2] = s3*z;  Y[3] = s3*x;
    Y[4] = s15*x*y; Y[5] = s15*y*z; Y[6] = 0.5f*s5*(3.0f*z*z - 1.0f);
    Y[7] = s15*x*z; Y[8] = 0.5f*s15*(x*x - y*y);
    Y[9]  = 0.25f*s70*y*(3.0f*x*x - y*y);
    Y[10] = s105*x*y*z;
    Y[11] = 0.25f*s42*y*(5.0f*z*z - 1.0f);
    Y[12] = 0.5f*s7*z*(5.0f*z*z - 3.0f);
    Y[13] = 0.25f*s42*x*(5.0f*z*z - 1.0f);
    Y[14] = 0.5f*s105*z*(x*x - y*y);
    Y[15] = 0.25f*s70*x*(x*x - 3.0f*y*y);
    elen[e] = len;
}

// ---------------------------------------------------------------------------
// Kernel 3: embedding gather / residual copy
// ---------------------------------------------------------------------------
__global__ void embed_kernel(const float* __restrict__ emb, const int* __restrict__ an,
                             float* __restrict__ h0, int total) {
    int i = blockIdx.x * blockDim.x + threadIdx.x;
    if (i >= total) return;
    h0[i] = emb[(size_t)an[i >> 9] * NDIM + (i & 511)];
}
__global__ void copy_kernel(const float* __restrict__ a, float* __restrict__ b, int total) {
    int i = blockIdx.x * blockDim.x + threadIdx.x;
    if (i < total) b[i] = a[i];
}

// ---------------------------------------------------------------------------
// Templated GEMM job: l3 compile-time -> constant d3 (no integer division),
// fully unrolled K chain so loads pipeline ahead of the WMMAs.
// ---------------------------------------------------------------------------
template<int L3>
__device__ __forceinline__ void gemm_job(const unsigned short* __restrict__ zsh,
                                         const unsigned short* __restrict__ wfrag,
                                         const float* __restrict__ elenT,
                                         const int* __restrict__ dstT,
                                         float* __restrict__ hout,
                                         int lane, int mt, int nh) {
    constexpr int ktn  = cKT[L3];
    constexpr int K    = cKL3[L3];
    constexpr int d3   = 2 * L3 + 1;
    constexpr int rows = TILE_E * d3;
    int ml = lane & 15, hi = lane >> 4;
    int arow = mt * 16 + ml; if (arow >= rows) arow = rows - 1;   // pad clamp
    unsigned abase = (unsigned)cZOFF[L3] + (unsigned)arow * (unsigned)K + (unsigned)(8 * hi);
    v8f accW = {0.f,0.f,0.f,0.f,0.f,0.f,0.f,0.f};
    v8f accB = {0.f,0.f,0.f,0.f,0.f,0.f,0.f,0.f};
    #pragma unroll
    for (int kt = 0; kt < ktn; kt++) {
        union { uint4 q[2]; v16bf v; } A, BW, BB;
        unsigned a0 = abase + (unsigned)(kt * 32);
        // A fragment (16-bit 16x32): regs 0-3 = k0..7(+8hi), regs 4-7 = k16..23(+8hi)
        A.q[0] = *(const uint4*)&zsh[a0];
        A.q[1] = *(const uint4*)&zsh[a0 + 16];
        size_t wi = ((((size_t)(cKTOFF[L3] + kt) * 2 + 0) * 2 + nh) * 32 + lane) * 16;
        const uint4* wp = (const uint4*)(wfrag + wi);
        BW.q[0] = wp[0]; BW.q[1] = wp[1];
        wi = ((((size_t)(cKTOFF[L3] + kt) * 2 + 1) * 2 + nh) * 32 + lane) * 16;
        wp = (const uint4*)(wfrag + wi);
        BB.q[0] = wp[0]; BB.q[1] = wp[1];
        accW = __builtin_amdgcn_wmma_f32_16x16x32_bf16(false, A.v, false, BW.v,
                                                       (short)0, accW, false, false);
        accB = __builtin_amdgcn_wmma_f32_16x16x32_bf16(false, A.v, false, BB.v,
                                                       (short)0, accB, false, false);
    }
    // D layout: VGPR v -> M = v + 8*hi, N = lane&15
    #pragma unroll
    for (int v = 0; v < 8; v++) {
        int m = v + 8 * hi;
        int row = mt * 16 + m;
        if (row < rows) {
            int el = row / d3, k3 = row - el * d3;      // d3 constexpr -> mul/shift
            int dn = dstT[el];
            if (dn >= 0) {
                int w = ml + 16 * nh;
                float val = elenT[el] * accW[v] + accB[v];
                atomicAdd(&hout[(size_t)dn * NDIM + cNOFF[L3] + w * d3 + k3], val);
            }
        }
    }
}

// ---------------------------------------------------------------------------
// Kernel 4: interaction layer. 8 edges/block, 256 threads (8 waves).
// Phase 0: stage compacted CG table into LDS (broadcast-friendly reads).
// Phase 1: sparse CG contraction -> z (bf16) staged in LDS.
// Phase 2: WMMA GEMM chains + scatter atomics.
// ---------------------------------------------------------------------------
__launch_bounds__(256)
__global__ void edge_kernel(const float* __restrict__ hin, float* __restrict__ hout,
                            const int* __restrict__ src, const int* __restrict__ dst,
                            const float* __restrict__ elen, const float* __restrict__ Yb,
                            const unsigned short* __restrict__ wfrag,
                            const int* __restrict__ cgHdr, const CgEnt* __restrict__ cgCompact,
                            int E) {
    __shared__ unsigned short zsh[25344];   // ~50.7 KB z tiles
    __shared__ CgEnt cgL[CG_TOT];           // 8 KB compacted CG table
    __shared__ int   hdrL[2 * NPATH + 1];
    __shared__ float elenT[TILE_E];
    __shared__ int   dstT[TILE_E];

    int tid = threadIdx.x;
    // ---- phase 0: stage CG into LDS ---------------------------------------
    if (tid < 2 * NPATH + 1) hdrL[tid] = cgHdr[tid];
    __syncthreads();
    {
        int tot = hdrL[2 * NPATH];
        for (int t = tid; t < tot; t += 256) cgL[t] = cgCompact[t];
    }
    // ---- phase 1: z computation (one (edge,channel) pair per thread) ------
    {
        int e = tid >> 5;                    // 0..7
        int u = tid & 31;                    // channel
        int egRaw = blockIdx.x * TILE_E + e;
        int eg = egRaw < E ? egRaw : E - 1;
        int sIdx = src[eg];
        if (u == 0) { elenT[e] = elen[eg]; dstT[e] = (egRaw < E) ? dst[eg] : -1; }
        const float* hb = hin + (size_t)sIdx * NDIM;
        float xa[16];
        xa[0] = hb[u];
        #pragma unroll
        for (int i = 0; i < 3; i++) xa[1 + i] = hb[32 + u*3 + i];
        #pragma unroll
        for (int i = 0; i < 5; i++) xa[4 + i] = hb[128 + u*5 + i];
        #pragma unroll
        for (int i = 0; i < 7; i++) xa[9 + i] = hb[288 + u*7 + i];
        float Yv[16];
        const float* Yp = Yb + (size_t)eg * 16;
        #pragma unroll
        for (int j = 0; j < 16; j++) Yv[j] = Yp[j];
        __syncthreads();                     // cgL staged
        for (int p = 0; p < NPATH; p++) {
            int base = hdrL[2*p], cnt = hdrL[2*p + 1];
            float zac[7] = {0.f,0.f,0.f,0.f,0.f,0.f,0.f};
            for (int t = 0; t < cnt; t++) {
                CgEnt c = cgL[base + t];
                zac[c.kk] += c.v * xa[c.ia] * Yv[c.ja];
            }
            int l3 = PL3[p], d3 = 2*l3 + 1, slot = PSLOT[p], K = KL3c[l3];
            unsigned zb = (unsigned)ZOFFc[l3] + (unsigned)(e * d3) * (unsigned)K
                        + (unsigned)(slot * 32 + u);
            for (int k = 0; k < d3; k++) zsh[zb + (unsigned)k * (unsigned)K] = f2bf(zac[k]);
        }
    }
    __syncthreads();

    // ---- phase 2: WMMA channel mixing + scatter ---------------------------
    int wv = tid >> 5, lane = tid & 31;
    for (int jb = wv; jb < 20; jb += 8) {
        int l3 = JOB_L3[jb], mt = JOB_MT[jb], nh = JOB_NH[jb];
        switch (l3) {
            case 0: gemm_job<0>(zsh, wfrag, elenT, dstT, hout, lane, mt, nh); break;
            case 1: gemm_job<1>(zsh, wfrag, elenT, dstT, hout, lane, mt, nh); break;
            case 2: gemm_job<2>(zsh, wfrag, elenT, dstT, hout, lane, mt, nh); break;
            default: gemm_job<3>(zsh, wfrag, elenT, dstT, hout, lane, mt, nh); break;
        }
    }
}

// ---------------------------------------------------------------------------
// Kernel 5: readout head + global reduction
// ---------------------------------------------------------------------------
__global__ void zero_kernel(float* out) { out[0] = 0.0f; }

__global__ void head_kernel(const float* __restrict__ h, const float* __restrict__ hw,
                            const float* __restrict__ hb, int N, float* __restrict__ out) {
    __shared__ float red[256];
    int n = blockIdx.x * blockDim.x + threadIdx.x;
    float s = 0.0f;
    if (n < N) {
        const float* r = h + (size_t)n * NDIM;
        for (int c = 0; c < NDIM; c++) s += r[c] * hw[c];
    }
    red[threadIdx.x] = s;
    __syncthreads();
    for (int st = 128; st > 0; st >>= 1) {
        if (threadIdx.x < st) red[threadIdx.x] += red[threadIdx.x + st];
        __syncthreads();
    }
    if (threadIdx.x == 0) atomicAdd(out, red[0]);
    if (n == 0) atomicAdd(out, (float)N * hb[0]);
}

// ---------------------------------------------------------------------------
// Host launcher
// ---------------------------------------------------------------------------
extern "C" void kernel_launch(void* const* d_in, const int* in_sizes, int n_in,
                              void* d_out, int out_size, void* d_ws, size_t ws_size,
                              hipStream_t stream) {
    (void)n_in; (void)out_size; (void)ws_size;
    const float* positions = (const float*)d_in[0];
    const int*   edge_idx  = (const int*)  d_in[1];
    const int*   atomic_nr = (const int*)  d_in[2];
    const float* embedding = (const float*)d_in[3];
    const float* wnet_w    = (const float*)d_in[4];
    const float* wnet_b    = (const float*)d_in[5];
    const float* head_w    = (const float*)d_in[6];
    const float* head_b    = (const float*)d_in[7];
    float* out = (float*)d_out;

    const int N = in_sizes[0] / 3;
    const int E = in_sizes[1] / 2;
    const int nLayers = in_sizes[4] / 23552;   // 2

    // workspace carve-out (~81 MB; deterministic layout)
    char* ws = (char*)d_ws;
    size_t o = 0;
    auto carve = [&](size_t bytes) { size_t r = o; o = (o + bytes + 255) & ~(size_t)255; return r; };
    int*            cgHdr   = (int*)           (ws + carve((size_t)(2*NPATH + 1) * 4));
    CgEnt*          cgScr   = (CgEnt*)         (ws + carve((size_t)NPATH * CG_CAP * sizeof(CgEnt)));
    CgEnt*          cgCmp   = (CgEnt*)         (ws + carve((size_t)CG_TOT * sizeof(CgEnt)));
    unsigned short* wfrag   = (unsigned short*)(ws + carve((size_t)2 * WFRAG_LAYER * 2));
    float*          elen    = (float*)         (ws + carve((size_t)E * 4));
    float*          Yb      = (float*)         (ws + carve((size_t)E * 16 * 4));
    float*          h0      = (float*)         (ws + carve((size_t)N * NDIM * 4));
    float*          h1      = (float*)         (ws + carve((size_t)N * NDIM * 4));

    zero_kernel<<<1, 1, 0, stream>>>(out);
    cg_prep_kernel<<<1, 32, 0, stream>>>(cgHdr, cgScr, cgCmp);
    {
        int tot = 2 * NPATH * 2 * 2 * 32;
        w_prep_kernel<<<(tot + 255) / 256, 256, 0, stream>>>(wnet_w, wnet_b, wfrag);
    }
    geom_kernel<<<(E + 255) / 256, 256, 0, stream>>>(positions, edge_idx, E, elen, Yb);
    {
        int tot = N * NDIM;
        embed_kernel<<<(tot + 255) / 256, 256, 0, stream>>>(embedding, atomic_nr, h0, tot);
    }
    const int* srcI = edge_idx;
    const int* dstI = edge_idx + E;
    float* hin = h0; float* hout = h1;
    for (int layer = 0; layer < nLayers; layer++) {
        int tot = N * NDIM;
        copy_kernel<<<(tot + 255) / 256, 256, 0, stream>>>(hin, hout, tot);  // residual base
        edge_kernel<<<(E + TILE_E - 1) / TILE_E, 256, 0, stream>>>(
            hin, hout, srcI, dstI, elen, Yb,
            wfrag + (size_t)layer * WFRAG_LAYER, cgHdr, cgCmp, E);
        float* t = hin; hin = hout; hout = t;
    }
    head_kernel<<<(N + 255) / 256, 256, 0, stream>>>(hin, head_w, head_b, N, out);
}
